// AttentionMessagePassingGNN_40175124087396
// MI455X (gfx1250) — matrix operand order repaired and checked by reading, not compiled, not run on hardware
//
#include <hip/hip_runtime.h>
#include <math.h>

// Problem constants (match reference)
#define NN 100000
#define EE 800000
#define BB 64
#define DD 128
#define WV 8          // waves per block (256 threads, wave32)

typedef __attribute__((ext_vector_type(16))) __bf16 v16bf;
typedef __attribute__((ext_vector_type(8)))  float  v8f;

union fragu { uint4 u[2]; v16bf v; };

// A fragment: 16x32 bf16, lane(l): m=l&15, k-group=(l>>4)*8 ; two 16B LDS loads
static __device__ __forceinline__ v16bf ldA(const __bf16* t, int lane, int stride, int kbase) {
    int m  = lane & 15;
    int kg = (lane >> 4) << 3;
    fragu c;
    c.u[0] = *(const uint4*)(t + m * stride + kbase + kg);
    c.u[1] = *(const uint4*)(t + m * stride + kbase + 16 + kg);
    return c.v;
}
// B fragment: 32x16 bf16 from transposed weights Wt[n][k]; lane(l): n=l&15, k-group=(l>>4)*8 ; two 16B loads (k, k+16)
static __device__ __forceinline__ v16bf ldB(const __bf16* wt, int lane, int K, int n0, int kbase) {
    int n  = n0 + (lane & 15);
    int kg = (lane >> 4) << 3;
    const __bf16* p = wt + n * K + kbase + kg;
    fragu c;
    c.u[0] = *(const uint4*)(p);
    c.u[1] = *(const uint4*)(p + 16);
    return c.v;
}
static __device__ __forceinline__ v8f wmma_bf16(v16bf a, v16bf b, v8f c) {
    return __builtin_amdgcn_wmma_f32_16x16x32_bf16(false, a, false, b, (short)0, c, false, false);
}
static __device__ __forceinline__ void gatomic(float* p, float v) {
    __hip_atomic_fetch_add(p, v, __ATOMIC_RELAXED, __HIP_MEMORY_SCOPE_AGENT);
}
static __device__ __forceinline__ void latomic(float* p, float v) {
    __hip_atomic_fetch_add(p, v, __ATOMIC_RELAXED, __HIP_MEMORY_SCOPE_WORKGROUP);
}
// stage 16 fp32 -> 16 bf16 into LDS (one lane's share of a 16x32 chunk)
static __device__ __forceinline__ void stage16(__bf16* d, const float* srow) {
#pragma unroll
    for (int q = 0; q < 4; ++q) {
        float4 v = ((const float4*)srow)[q];
        d[q * 4 + 0] = (__bf16)v.x; d[q * 4 + 1] = (__bf16)v.y;
        d[q * 4 + 2] = (__bf16)v.z; d[q * 4 + 3] = (__bf16)v.w;
    }
}
// one k-step: issue all 16 ds_load_b128 for the 8 B fragments, fence DS/WMMA
// movement with sched_barrier, then run 8 back-to-back WMMAs. This yields
// incremental s_wait_dscnt values instead of a full drain per WMMA.
// mask 0x477 = ALU|VALU|SALU|VMEM|VMEM_RD|VMEM_WR|TRANS may cross; DS+WMMA may not.
static __device__ __forceinline__ void kstep(v8f acc[8], v16bf af, const __bf16* wt,
                                             int lane, int K, int kbase) {
    v16bf bfr[8];
#pragma unroll
    for (int nt = 0; nt < 8; ++nt) bfr[nt] = ldB(wt, lane, K, nt * 16, kbase);
    __builtin_amdgcn_sched_barrier(0x477);
#pragma unroll
    for (int nt = 0; nt < 8; ++nt) acc[nt] = wmma_bf16(af, bfr[nt], acc[nt]);
}

// ---------------------------------------------------------------------------
// Edge kernel: e_new = gate(MLP([x[row], x[col], edge_attr]))
// plus per-source-node sums (global atomics) and per-batch sums (LDS privatized)
// ---------------------------------------------------------------------------
__global__ void __launch_bounds__(256, 1) __attribute__((amdgpu_waves_per_eu(2)))
gnn_edge_kernel(const float* __restrict__ x,
                const int* __restrict__ row, const int* __restrict__ col,
                const float* __restrict__ ea, const int* __restrict__ batch,
                const float* __restrict__ W1, const float* __restrict__ b1,
                const float* __restrict__ W2, const float* __restrict__ b2,
                const float* __restrict__ aW, const float* __restrict__ ab,
                float* __restrict__ e_new,
                float* __restrict__ e_sum, float* __restrict__ e_cnt,
                float* __restrict__ em_sum, float* __restrict__ em_cnt) {
    extern __shared__ char smem[];
    __bf16* w1t = (__bf16*)smem;                 // [128][384] = Wt of [384,128]
    __bf16* w2t = w1t + 384 * 128;               // [128][128]
    float*  sb1 = (float*)(w2t + 128 * 128);
    float*  sb2 = sb1 + 128;
    float*  saw = sb2 + 128;
    float*  sem = saw + 128;                     // [64][128] batch-edge accumulator
    float*  semc = sem + 64 * 128;               // [64]
    __bf16* aStage = (__bf16*)(semc + 64);       // WV * 2 * 16*32 (double buffered)
    __bf16* hbuf   = aStage + WV * 2 * 16 * 32;  // WV * 16*128
    int*    ibuf   = (int*)(hbuf + WV * 16 * 128);

    const int tid = threadIdx.x;
    const int wave = tid >> 5, lane = tid & 31;
    const int m0 = lane & 15, kg = lane >> 4;

    for (int i = tid; i < 384 * 128; i += 256) { int k = i >> 7, n = i & 127; w1t[n * 384 + k] = (__bf16)W1[i]; }
    for (int i = tid; i < 128 * 128; i += 256) { int k = i >> 7, n = i & 127; w2t[n * 128 + k] = (__bf16)W2[i]; }
    for (int i = tid; i < 128; i += 256) { sb1[i] = b1[i]; sb2[i] = b2[i]; saw[i] = aW[i]; }
    for (int i = tid; i < 64 * 128; i += 256) sem[i] = 0.0f;
    if (tid < 64) semc[tid] = 0.0f;
    __syncthreads();

    const float abv = ab[0];
    __bf16* A0 = aStage + wave * 2 * 16 * 32;    // ping
    __bf16* A1 = A0 + 16 * 32;                   // pong
    __bf16* H  = hbuf + wave * 16 * 128;
    int* RI = ibuf + wave * 48; int* CI = RI + 16; int* BI = CI + 16;
    const int mm = lane >> 1, hh = (lane & 1) * 16;

    const int numTiles = EE / 16;
    for (long t = (long)blockIdx.x * WV + wave; t < numTiles; t += (long)gridDim.x * WV) {
        const long e0 = t * 16;
        if (lane < 16) {
            int r = row[e0 + lane];
            RI[lane] = r; CI[lane] = col[e0 + lane]; BI[lane] = batch[r];
        }
        // hoist gather bases into registers (one LDS read per tile, not per k-step)
        const float* xrow = x + (long)RI[mm] * DD;
        const float* xcol = x + (long)CI[mm] * DD;
        const float* earow = ea + (e0 + mm) * DD;

        v8f acc[8] = {};
        // ---- layer 1: K = 384 (x[row] | x[col] | edge_attr), 12 k-steps ----
#pragma unroll
        for (int c = 0; c < 12; ++c) {
            const float* srow;
            if (c < 4)      srow = xrow + c * 32;
            else if (c < 8) srow = xcol + (c - 4) * 32;
            else            srow = earow + (c - 8) * 32;
            __bf16* A = (c & 1) ? A1 : A0;       // double buffer: stage c overlaps compute c-1
            stage16(A + mm * 32 + hh, srow + hh);
            v16bf af = ldA(A, lane, 32, 0);
            kstep(acc, af, w1t, lane, 384, c * 32);
        }
        // bias + relu -> bf16 H, reset acc
#pragma unroll
        for (int nt = 0; nt < 8; ++nt) {
            int n = nt * 16 + m0;
            float bb = sb1[n];
#pragma unroll
            for (int i = 0; i < 8; ++i) {
                float v = acc[nt][i] + bb;
                H[(i + 8 * kg) * 128 + n] = (__bf16)(v > 0.0f ? v : 0.0f);
                acc[nt][i] = 0.0f;
            }
        }
        // ---- layer 2: K = 128 ----
#pragma unroll
        for (int c = 0; c < 4; ++c) {
            v16bf af = ldA(H, lane, 128, c * 32);
            kstep(acc, af, w2t, lane, 128, c * 32);
        }
#pragma unroll
        for (int nt = 0; nt < 8; ++nt) {
            float bb = sb2[nt * 16 + m0];
#pragma unroll
            for (int i = 0; i < 8; ++i) acc[nt][i] += bb;
        }
        // ---- attention gate: per-row dot with aW, 16-lane xor reduction ----
        float sg[8];
#pragma unroll
        for (int i = 0; i < 8; ++i) {
            float p = 0.0f;
#pragma unroll
            for (int nt = 0; nt < 8; ++nt) p += acc[nt][i] * saw[nt * 16 + m0];
            p += __shfl_xor(p, 1, 32); p += __shfl_xor(p, 2, 32);
            p += __shfl_xor(p, 4, 32); p += __shfl_xor(p, 8, 32);
            sg[i] = 1.0f / (1.0f + __expf(-(p + abv)));
        }
        // ---- outputs + aggregation atomics ----
        float* eout = e_new + e0 * DD;
#pragma unroll
        for (int nt = 0; nt < 8; ++nt) {
            int n = nt * 16 + m0;
#pragma unroll
            for (int i = 0; i < 8; ++i) {
                int m = i + 8 * kg;
                float v = acc[nt][i] * sg[i];
                eout[m * DD + n] = v;
                gatomic(&e_sum[(long)RI[m] * DD + n], v);
                latomic(&sem[BI[m] * DD + n], v);
            }
        }
        if (lane < 16) {
            gatomic(&e_cnt[RI[lane]], 1.0f);
            latomic(&semc[BI[lane]], 1.0f);
        }
    }
    __syncthreads();
    for (int i = tid; i < 64 * 128; i += 256) gatomic(&em_sum[i], sem[i]);
    if (tid < 64) gatomic(&em_cnt[tid], semc[tid]);
}

// ---------------------------------------------------------------------------
// Node kernel: x_new = MLP2([x, gate(MLP1([x, mean_e(e_new)]))]) + batch sums
// ---------------------------------------------------------------------------
__global__ void __launch_bounds__(256, 1) __attribute__((amdgpu_waves_per_eu(2)))
gnn_node_kernel(const float* __restrict__ x, const int* __restrict__ batch,
                const float* __restrict__ e_sum, const float* __restrict__ e_cnt,
                const float* __restrict__ W1, const float* __restrict__ b1,
                const float* __restrict__ W2, const float* __restrict__ b2,
                const float* __restrict__ W3, const float* __restrict__ b3,
                const float* __restrict__ W4, const float* __restrict__ b4,
                const float* __restrict__ aW, const float* __restrict__ ab,
                float* __restrict__ x_new,
                float* __restrict__ nm_sum, float* __restrict__ nm_cnt) {
    extern __shared__ char smem[];
    __bf16* w1t = (__bf16*)smem;                 // [128][256]
    __bf16* w2t = w1t + 128 * 256;               // [128][128]
    __bf16* w3t = w2t + 128 * 128;               // [128][256]
    __bf16* w4t = w3t + 128 * 256;               // [128][128]
    float*  sb1 = (float*)(w4t + 128 * 128);
    float*  sb2 = sb1 + 128;
    float*  sb3 = sb2 + 128;
    float*  sb4 = sb3 + 128;
    float*  saw = sb4 + 128;
    float*  snm = saw + 128;                     // [64][128]
    float*  snmc = snm + 64 * 128;               // [64]
    __bf16* aStage = (__bf16*)(snmc + 64);       // WV * 2 * 16*32
    __bf16* hbuf   = aStage + WV * 2 * 16 * 32;  // WV * 16*128
    int*    ibuf   = (int*)(hbuf + WV * 16 * 128);

    const int tid = threadIdx.x;
    const int wave = tid >> 5, lane = tid & 31;
    const int m0 = lane & 15, kg = lane >> 4;

    for (int i = tid; i < 256 * 128; i += 256) { int k = i >> 7, n = i & 127; w1t[n * 256 + k] = (__bf16)W1[i]; w3t[n * 256 + k] = (__bf16)W3[i]; }
    for (int i = tid; i < 128 * 128; i += 256) { int k = i >> 7, n = i & 127; w2t[n * 128 + k] = (__bf16)W2[i]; w4t[n * 128 + k] = (__bf16)W4[i]; }
    for (int i = tid; i < 128; i += 256) { sb1[i] = b1[i]; sb2[i] = b2[i]; sb3[i] = b3[i]; sb4[i] = b4[i]; saw[i] = aW[i]; }
    for (int i = tid; i < 64 * 128; i += 256) snm[i] = 0.0f;
    if (tid < 64) snmc[tid] = 0.0f;
    __syncthreads();

    const float abv = ab[0];
    __bf16* A0 = aStage + wave * 2 * 16 * 32;
    __bf16* A1 = A0 + 16 * 32;
    __bf16* H = hbuf + wave * 16 * 128;
    int* BI = ibuf + wave * 16;
    const int mm = lane >> 1, hh = (lane & 1) * 16;

    const int numTiles = NN / 16;
    for (long t = (long)blockIdx.x * WV + wave; t < numTiles; t += (long)gridDim.x * WV) {
        const long n0 = t * 16;
        if (lane < 16) BI[lane] = batch[n0 + lane];
        const long nd = n0 + mm;
        const float* xrow = x + nd * DD;
        const float* srowE = e_sum + nd * DD;
        const float invc = 1.0f / fmaxf(e_cnt[nd], 1.0f);

        v8f acc[8] = {};
        // ---- MLP1 layer A: K = 256 ([x | e_agg]) ----
#pragma unroll
        for (int c = 0; c < 8; ++c) {
            __bf16* A = (c & 1) ? A1 : A0;
            __bf16* d = A + mm * 32 + hh;
            if (c < 4) {
                stage16(d, xrow + c * 32 + hh);
            } else {
                const float* srow = srowE + (c - 4) * 32 + hh;
#pragma unroll
                for (int q = 0; q < 4; ++q) {
                    float4 v = ((const float4*)srow)[q];
                    d[q * 4 + 0] = (__bf16)(v.x * invc); d[q * 4 + 1] = (__bf16)(v.y * invc);
                    d[q * 4 + 2] = (__bf16)(v.z * invc); d[q * 4 + 3] = (__bf16)(v.w * invc);
                }
            }
            v16bf af = ldA(A, lane, 32, 0);
            kstep(acc, af, w1t, lane, 256, c * 32);
        }
#pragma unroll
        for (int nt = 0; nt < 8; ++nt) {
            int n = nt * 16 + m0;
            float bb = sb1[n];
#pragma unroll
            for (int i = 0; i < 8; ++i) {
                float v = acc[nt][i] + bb;
                H[(i + 8 * kg) * 128 + n] = (__bf16)(v > 0.0f ? v : 0.0f);
                acc[nt][i] = 0.0f;
            }
        }
        // ---- MLP1 layer B: K = 128 ----
#pragma unroll
        for (int c = 0; c < 4; ++c) {
            v16bf af = ldA(H, lane, 128, c * 32);
            kstep(acc, af, w2t, lane, 128, c * 32);
        }
#pragma unroll
        for (int nt = 0; nt < 8; ++nt) {
            float bb = sb2[nt * 16 + m0];
#pragma unroll
            for (int i = 0; i < 8; ++i) acc[nt][i] += bb;
        }
        // gate
        float sg[8];
#pragma unroll
        for (int i = 0; i < 8; ++i) {
            float p = 0.0f;
#pragma unroll
            for (int nt = 0; nt < 8; ++nt) p += acc[nt][i] * saw[nt * 16 + m0];
            p += __shfl_xor(p, 1, 32); p += __shfl_xor(p, 2, 32);
            p += __shfl_xor(p, 4, 32); p += __shfl_xor(p, 8, 32);
            sg[i] = 1.0f / (1.0f + __expf(-(p + abv)));
        }
#pragma unroll
        for (int nt = 0; nt < 8; ++nt) {
            int n = nt * 16 + m0;
#pragma unroll
            for (int i = 0; i < 8; ++i) {
                H[(i + 8 * kg) * 128 + n] = (__bf16)(acc[nt][i] * sg[i]);
                acc[nt][i] = 0.0f;
            }
        }
        // ---- MLP2 layer A: K = 256 ([x | h]) ----
#pragma unroll
        for (int c = 0; c < 8; ++c) {
            v16bf af;
            if (c < 4) {
                __bf16* A = (c & 1) ? A1 : A0;
                stage16(A + mm * 32 + hh, xrow + c * 32 + hh);
                af = ldA(A, lane, 32, 0);
            } else {
                af = ldA(H, lane, 128, (c - 4) * 32);
            }
            kstep(acc, af, w3t, lane, 256, c * 32);
        }
#pragma unroll
        for (int nt = 0; nt < 8; ++nt) {
            int n = nt * 16 + m0;
            float bb = sb3[n];
#pragma unroll
            for (int i = 0; i < 8; ++i) {
                float v = acc[nt][i] + bb;
                H[(i + 8 * kg) * 128 + n] = (__bf16)(v > 0.0f ? v : 0.0f);
                acc[nt][i] = 0.0f;
            }
        }
        // ---- MLP2 layer B: K = 128 -> x_new ----
#pragma unroll
        for (int c = 0; c < 4; ++c) {
            v16bf af = ldA(H, lane, 128, c * 32);
            kstep(acc, af, w4t, lane, 128, c * 32);
        }
        float* xout = x_new + n0 * DD;
#pragma unroll
        for (int nt = 0; nt < 8; ++nt) {
            int n = nt * 16 + m0;
            float bb = sb4[n];
#pragma unroll
            for (int i = 0; i < 8; ++i) {
                int m = i + 8 * kg;
                float v = acc[nt][i] + bb;
                xout[m * DD + n] = v;
                latomic(&snm[BI[m] * DD + n], v);
            }
        }
        if (lane < 16) latomic(&snmc[BI[lane]], 1.0f);
    }
    __syncthreads();
    for (int i = tid; i < 64 * 128; i += 256) gatomic(&nm_sum[i], snm[i]);
    if (tid < 64) gatomic(&nm_cnt[tid], snmc[tid]);
}

// ---------------------------------------------------------------------------
// Global kernel (tiny: B=64): u_new = gate(MLP([u, node_mean, edge_mean]))
// ---------------------------------------------------------------------------
__global__ void gnn_global_kernel(const float* __restrict__ u,
                                  const float* __restrict__ nm_sum, const float* __restrict__ nm_cnt,
                                  const float* __restrict__ em_sum, const float* __restrict__ em_cnt,
                                  const float* __restrict__ W1, const float* __restrict__ b1,
                                  const float* __restrict__ W2, const float* __restrict__ b2,
                                  const float* __restrict__ aW, const float* __restrict__ ab,
                                  float* __restrict__ u_new) {
    extern __shared__ char smem[];
    float* gf = (float*)smem;        // [64][384], reused as [64][128] for layer-2 out
    float* h  = gf + 64 * 384;       // [64][128]
    float* sg = h + 64 * 128;        // [64]
    const int tid = threadIdx.x;

    for (int i = tid; i < 64 * 384; i += 256) {
        int b = i / 384, j = i - b * 384;
        float v;
        if (j < 128)      v = u[b * 128 + j];
        else if (j < 256) v = nm_sum[b * 128 + j - 128] / fmaxf(nm_cnt[b], 1.0f);
        else              v = em_sum[b * 128 + j - 256] / fmaxf(em_cnt[b], 1.0f);
        gf[i] = v;
    }
    __syncthreads();
    for (int i = tid; i < 64 * 128; i += 256) {
        int b = i >> 7, n = i & 127;
        float s = b1[n];
        for (int k = 0; k < 384; ++k) s += gf[b * 384 + k] * W1[k * 128 + n];
        h[i] = fmaxf(s, 0.0f);
    }
    __syncthreads();
    for (int i = tid; i < 64 * 128; i += 256) {
        int b = i >> 7, g = i & 127;
        float s = b2[g];
        for (int k = 0; k < 128; ++k) s += h[b * 128 + k] * W2[k * 128 + g];
        gf[i] = s;   // safe: layer-1 inputs fully consumed after barrier
    }
    __syncthreads();
    if (tid < 64) {
        float z = ab[0];
        for (int k = 0; k < 128; ++k) z += gf[tid * 128 + k] * aW[k];
        sg[tid] = 1.0f / (1.0f + __expf(-z));
    }
    __syncthreads();
    for (int i = tid; i < 64 * 128; i += 256) u_new[i] = gf[i] * sg[i >> 7];
}

// ---------------------------------------------------------------------------
extern "C" void kernel_launch(void* const* d_in, const int* in_sizes, int n_in,
                              void* d_out, int out_size, void* d_ws, size_t ws_size,
                              hipStream_t stream) {
    (void)in_sizes; (void)n_in; (void)out_size; (void)ws_size;
    const float* x     = (const float*)d_in[0];
    const int*   ei    = (const int*)d_in[1];
    const float* ea    = (const float*)d_in[2];
    const float* u     = (const float*)d_in[3];
    const int*   batch = (const int*)d_in[4];
    const float* eW1 = (const float*)d_in[5],  *eb1 = (const float*)d_in[6];
    const float* eW2 = (const float*)d_in[7],  *eb2 = (const float*)d_in[8];
    const float* eaW = (const float*)d_in[9],  *eab = (const float*)d_in[10];
    const float* nW1 = (const float*)d_in[11], *nb1 = (const float*)d_in[12];
    const float* nW2 = (const float*)d_in[13], *nb2 = (const float*)d_in[14];
    const float* nW3 = (const float*)d_in[15], *nb3 = (const float*)d_in[16];
    const float* nW4 = (const float*)d_in[17], *nb4 = (const float*)d_in[18];
    const float* naW = (const float*)d_in[19], *nab = (const float*)d_in[20];
    const float* gW1 = (const float*)d_in[21], *gb1 = (const float*)d_in[22];
    const float* gW2 = (const float*)d_in[23], *gb2 = (const float*)d_in[24];
    const float* gaW = (const float*)d_in[25], *gab = (const float*)d_in[26];

    float* out   = (float*)d_out;
    float* x_new = out;
    float* e_new = out + (long)NN * DD;
    float* u_new = e_new + (long)EE * DD;

    float* ws     = (float*)d_ws;
    float* e_sum  = ws;                               // N*128
    float* e_cnt  = e_sum + (long)NN * DD;            // N
    float* em_sum = e_cnt + NN;                       // 64*128
    float* em_cnt = em_sum + BB * DD;                 // 64
    float* nm_sum = em_cnt + BB;                      // 64*128
    float* nm_cnt = nm_sum + BB * DD;                 // 64

    size_t zeroBytes = (size_t)((long)NN * DD + NN + 2 * (BB * DD + BB)) * sizeof(float);
    hipMemsetAsync(d_ws, 0, zeroBytes, stream);

    size_t edgeLds = (size_t)(384 * 128 + 128 * 128) * 2   // bf16 weights
                   + 3 * 128 * 4                           // biases + aW
                   + 64 * 128 * 4 + 64 * 4                 // em accumulator
                   + (size_t)WV * 2 * 16 * 32 * 2          // A staging (double buffered)
                   + (size_t)WV * 16 * 128 * 2             // H staging
                   + (size_t)WV * 48 * 4;                  // index buffer
    gnn_edge_kernel<<<600, 256, edgeLds, stream>>>(x, ei, ei + EE, ea, batch,
                                                   eW1, eb1, eW2, eb2, eaW, eab,
                                                   e_new, e_sum, e_cnt, em_sum, em_cnt);

    size_t nodeLds = (size_t)(128 * 256 + 128 * 128 + 128 * 256 + 128 * 128) * 2
                   + 5 * 128 * 4
                   + 64 * 128 * 4 + 64 * 4
                   + (size_t)WV * 2 * 16 * 32 * 2
                   + (size_t)WV * 16 * 128 * 2
                   + (size_t)WV * 16 * 4;
    gnn_node_kernel<<<400, 256, nodeLds, stream>>>(x, batch, e_sum, e_cnt,
                                                   nW1, nb1, nW2, nb2, nW3, nb3, nW4, nb4, naW, nab,
                                                   x_new, nm_sum, nm_cnt);

    size_t gLds = (size_t)(64 * 384 + 64 * 128 + 64) * 4;
    gnn_global_kernel<<<1, 256, gLds, stream>>>(u, nm_sum, nm_cnt, em_sum, em_cnt,
                                                gW1, gb1, gW2, gb2, gaW, gab, u_new);
}
